// DynamicKGE_62818191671725
// MI455X (gfx1250) — compile-verified, matrix operand order checked by reading, not run on hardware
//
#include <hip/hip_runtime.h>
#include <hip/hip_bf16.h>
#include <math.h>

// ---------------------------------------------------------------------------
// DynamicKGE forward for MI455X (gfx1250, wave32, WMMA).
//
//   cvt:  X0 (f32) -> X0b (bf16, row-major)          [once per call]
//   L1:   Y = X0b @ [basis1_0..3 | root1]  (20 static v_wmma per wave)
//         edge combine + mean scatter (global f32 atomics, Y L2-resident)
//         finalize -> X2b (bf16, feeds layer-2 GEMM directly)
//   L2:   Y = X2b @ [basis2_0..3 | root2]; scatter; finalize -> X3 (f32)
//   rel:  relu(DAD @ Rtab @ Wrel) -> RC
//   out:  gated gather -> d_out [3, 20000, 128]
// ---------------------------------------------------------------------------

typedef __attribute__((ext_vector_type(16))) __bf16 v16bf;
typedef __attribute__((ext_vector_type(8)))  __bf16 v8bf;
typedef __attribute__((ext_vector_type(8)))  float  v8f;

#define KN_ENT   50000
#define KDIM     128
#define KNREL    200
#define KNE      200000
#define KNS      20000
#define KWCOLS   640      // 4 bases * 128 + root 128
#define KNCT     40       // column tiles of 16
#define KMT      3125     // row tiles of 16 (50000 / 16 exact)
#define KKT      4        // k tiles of 32 (128 / 32)
#define KNTG     8        // column-tile groups per row tile
#define KNTW     5        // column tiles per wave (KNTG * KNTW == KNCT)

__device__ __forceinline__ __bf16 f2bf(float f) {
  unsigned u = __builtin_bit_cast(unsigned, f);
  unsigned r = u + 0x7FFFu + ((u >> 16) & 1u);   // round-to-nearest-even
  unsigned short h = (unsigned short)(r >> 16);
  return __builtin_bit_cast(__bf16, h);
}

__device__ __forceinline__ float sigm(float x) { return 1.0f / (1.0f + __expf(-x)); }

__device__ __forceinline__ void atomic_fadd(float* p, float v) {
  __hip_atomic_fetch_add(p, v, __ATOMIC_RELAXED, __HIP_MEMORY_SCOPE_AGENT);
}

// ---------------- zero fill ------------------------------------------------
__global__ void dkge_zero(float* __restrict__ p, int n) {
  int i = blockIdx.x * blockDim.x + threadIdx.x;
  if (i < n) p[i] = 0.0f;
}

// ---------------- f32 -> bf16 row-major convert ----------------------------
__global__ void dkge_cvt_bf16(const float* __restrict__ in, __bf16* __restrict__ out, int n) {
  int i = blockIdx.x * blockDim.x + threadIdx.x;
  if (i < n) out[i] = f2bf(in[i]);
}

// ---------------- pack weights into WMMA B-fragment order (bf16) -----------
// Wpack index: (((kt*KNCT + nt)*32 + lane)*16 + i)
// B fragment (32x16 bf16, K x N): lane -> N = lane&15, half = lane>>4;
// element i -> K = (i<8 ? half*8+i : 16 + half*8 + (i-8)) within the k-tile.
__global__ void dkge_pack_w(const float* __restrict__ basis,  // [4,128,128]
                            const float* __restrict__ root,   // [128,128]
                            __bf16* __restrict__ Wpack) {
  int idx = blockIdx.x * blockDim.x + threadIdx.x;
  if (idx >= KKT * KNCT * 32 * 16) return;
  int i    = idx & 15;
  int lane = (idx >> 4) & 31;
  int nt   = (idx >> 9) % KNCT;
  int kt   = idx / (KNCT * 512);
  int half = lane >> 4;
  int kin  = (i < 8) ? (half * 8 + i) : (16 + half * 8 + (i - 8));
  int K    = kt * 32 + kin;
  int col  = nt * 16 + (lane & 15);
  float w;
  if (col < 512) {                 // basis block b, out column o
    int b = col >> 7, o = col & 127;
    w = basis[(b * KDIM + K) * KDIM + o];
  } else {                         // root block
    w = root[K * KDIM + (col - 512)];
  }
  Wpack[idx] = f2bf(w);
}

// ---------------- bf16 WMMA GEMM: Y[50000 x 640] = Xb @ Wcat ---------------
// One wave per (row tile, group of 5 column tiles): A fragments reused 5x,
// 20 static v_wmma per wave, exact launch so EXEC is all-ones.
__global__ void dkge_gemm_wmma(const __bf16* __restrict__ Xb,   // [N,128] bf16
                               const __bf16* __restrict__ Wpack,
                               float* __restrict__ Y) {
  int gid   = blockIdx.x * blockDim.x + threadIdx.x;
  int wave  = gid >> 5;
  int lane  = threadIdx.x & 31;
  int mTile = wave / KNTG;
  int nt0   = (wave % KNTG) * KNTW;
  int half  = lane >> 4;
  int l15   = lane & 15;

  // A fragment source: row (mTile*16 + l15), runs of 8 K at half*8 and 16+half*8
  const __bf16* ar = Xb + (mTile * 16 + l15) * KDIM;

  v8f acc[KNTW];
  #pragma unroll
  for (int j = 0; j < KNTW; ++j) acc[j] = (v8f){};

  #pragma unroll
  for (int kt = 0; kt < KKT; ++kt) {
    int kbase = kt * 32 + half * 8;
    v8bf alo = *(const v8bf*)(ar + kbase);        // 16B aligned b128 load
    v8bf ahi = *(const v8bf*)(ar + kbase + 16);
    v16bf a = __builtin_shufflevector(alo, ahi,
        0, 1, 2, 3, 4, 5, 6, 7, 8, 9, 10, 11, 12, 13, 14, 15);

    const __bf16* wk = Wpack + ((kt * KNCT + nt0) * 32 + lane) * 16;
    #pragma unroll
    for (int j = 0; j < KNTW; ++j) {
      v16bf b = *(const v16bf*)(wk + j * (32 * 16));
      acc[j] = __builtin_amdgcn_wmma_f32_16x16x32_bf16(
          false, a, false, b, (short)0, acc[j], false, false);
    }
  }

  // C layout: VGPR r -> row M = r + 8*half, col N = lane&15
  int rowBase = mTile * 16 + half * 8;
  #pragma unroll
  for (int j = 0; j < KNTW; ++j) {
    int col = (nt0 + j) * 16 + l15;
    #pragma unroll
    for (int r = 0; r < 8; ++r)
      Y[(rowBase + r) * KWCOLS + col] = acc[j][r];
  }
}

// ---------------- edge combine + mean-scatter ------------------------------
// One wave per edge; lane owns 4 contiguous columns (128/32).
__global__ void dkge_edge_scatter(const int* __restrict__ ei,     // [2,E]
                                  const int* __restrict__ etype,  // [E]
                                  const float* __restrict__ enorm,// [E]
                                  const float* __restrict__ att,  // [400,4]
                                  const float* __restrict__ Y,    // [N,640]
                                  float* __restrict__ agg,        // [N,128]
                                  float* __restrict__ cnt) {      // [N]
  int gid  = blockIdx.x * blockDim.x + threadIdx.x;
  int e    = gid >> 5;
  int lane = gid & 31;
  if (e >= KNE) return;
  int src = ei[e];
  int dst = ei[KNE + e];
  int t   = etype[e];
  float nrm = enorm[e];
  float a0 = att[t * 4 + 0] * nrm;
  float a1 = att[t * 4 + 1] * nrm;
  float a2 = att[t * 4 + 2] * nrm;
  float a3 = att[t * 4 + 3] * nrm;

  const float* ys = Y + src * KWCOLS;
  int c = lane * 4;
  float4 y0 = *(const float4*)(ys + 0   + c);
  float4 y1 = *(const float4*)(ys + 128 + c);
  float4 y2 = *(const float4*)(ys + 256 + c);
  float4 y3 = *(const float4*)(ys + 384 + c);

  float m0 = a0 * y0.x + a1 * y1.x + a2 * y2.x + a3 * y3.x;
  float m1 = a0 * y0.y + a1 * y1.y + a2 * y2.y + a3 * y3.y;
  float m2 = a0 * y0.z + a1 * y1.z + a2 * y2.z + a3 * y3.z;
  float m3 = a0 * y0.w + a1 * y1.w + a2 * y2.w + a3 * y3.w;

  float* ag = agg + dst * KDIM + c;
  atomic_fadd(ag + 0, m0);
  atomic_fadd(ag + 1, m1);
  atomic_fadd(ag + 2, m2);
  atomic_fadd(ag + 3, m3);
  if (lane == 0) atomic_fadd(&cnt[dst], 1.0f);
}

// ---------------- finalize: mean + root + bias + relu ----------------------
// Layer 1 variant: emit bf16 (feeds the next WMMA GEMM directly).
__global__ void dkge_finalize_bf16(const float* __restrict__ agg,
                                   const float* __restrict__ cnt,
                                   const float* __restrict__ Y,
                                   const float* __restrict__ bias,
                                   __bf16* __restrict__ Xout) {
  int idx = blockIdx.x * blockDim.x + threadIdx.x;
  if (idx >= KN_ENT * KDIM) return;
  int node = idx >> 7;
  int c    = idx & 127;
  float d  = cnt[node];
  d = d > 1.0f ? d : 1.0f;
  float v = agg[idx] / d + Y[node * KWCOLS + 512 + c] + bias[c];
  Xout[idx] = f2bf(v > 0.0f ? v : 0.0f);
}

// Layer 2 variant: emit f32 (consumed by the gather epilogue).
__global__ void dkge_finalize_f32(const float* __restrict__ agg,
                                  const float* __restrict__ cnt,
                                  const float* __restrict__ Y,
                                  const float* __restrict__ bias,
                                  float* __restrict__ Xout) {
  int idx = blockIdx.x * blockDim.x + threadIdx.x;
  if (idx >= KN_ENT * KDIM) return;
  int node = idx >> 7;
  int c    = idx & 127;
  float d  = cnt[node];
  d = d > 1.0f ? d : 1.0f;
  float v = agg[idx] / d + Y[node * KWCOLS + 512 + c] + bias[c];
  Xout[idx] = v > 0.0f ? v : 0.0f;
}

// ---------------- relation GCN (tiny) --------------------------------------
__global__ void dkge_rel_stage1(const float* __restrict__ DAD,   // [200,200]
                                const float* __restrict__ Rtab,  // [200,128]
                                float* __restrict__ T1) {        // [200,128]
  int idx = blockIdx.x * blockDim.x + threadIdx.x;
  if (idx >= KNREL * KDIM) return;
  int i = idx >> 7, k = idx & 127;
  float s = 0.0f;
  for (int j = 0; j < KNREL; ++j) s += DAD[i * KNREL + j] * Rtab[j * KDIM + k];
  T1[idx] = s;
}

__global__ void dkge_rel_stage2(const float* __restrict__ T1,    // [200,128]
                                const float* __restrict__ W,     // [128,128]
                                float* __restrict__ RC) {        // [200,128]
  int idx = blockIdx.x * blockDim.x + threadIdx.x;
  if (idx >= KNREL * KDIM) return;
  int i = idx >> 7, o = idx & 127;
  float s = 0.0f;
  for (int k = 0; k < KDIM; ++k) s += T1[i * KDIM + k] * W[k * KDIM + o];
  RC[idx] = s > 0.0f ? s : 0.0f;
}

// ---------------- gated gather output --------------------------------------
__global__ void dkge_output(const int* __restrict__ samples,      // [S,3]
                            const float* __restrict__ ent_emb,    // [N,128]
                            const float* __restrict__ rel_emb,    // [200,128]
                            const float* __restrict__ ent_ctx,    // [N,128]
                            const float* __restrict__ rel_ctx,    // [200,128]
                            const float* __restrict__ gate_e,     // [128]
                            const float* __restrict__ gate_r,     // [128]
                            float* __restrict__ out) {            // [3,S,128]
  int idx = blockIdx.x * blockDim.x + threadIdx.x;
  if (idx >= KNS * KDIM) return;
  int s = idx >> 7;
  int c = idx & 127;
  int h = samples[s * 3 + 0];
  int r = samples[s * 3 + 1];
  int t = samples[s * 3 + 2];
  float ge = sigm(gate_e[c]);
  float gr = sigm(gate_r[c]);
  out[0 * KNS * KDIM + idx] = ge * ent_emb[h * KDIM + c] + (1.0f - ge) * ent_ctx[h * KDIM + c];
  out[1 * KNS * KDIM + idx] = gr * rel_emb[r * KDIM + c] + (1.0f - gr) * rel_ctx[r * KDIM + c];
  out[2 * KNS * KDIM + idx] = ge * ent_emb[t * KDIM + c] + (1.0f - ge) * ent_ctx[t * KDIM + c];
}

// ---------------------------------------------------------------------------
extern "C" void kernel_launch(void* const* d_in, const int* in_sizes, int n_in,
                              void* d_out, int out_size, void* d_ws, size_t ws_size,
                              hipStream_t stream) {
  // Inputs (setup_inputs order)
  // 0 entity (arange, gather is identity)
  const int*   edge_index = (const int*)  d_in[1];
  const int*   edge_type  = (const int*)  d_in[2];
  const float* edge_norm  = (const float*)d_in[3];
  const int*   samples    = (const int*)  d_in[4];
  const float* DAD_rel    = (const float*)d_in[5];
  const float* ent_emb    = (const float*)d_in[6];
  const float* rel_emb    = (const float*)d_in[7];
  const float* ent_ctx0   = (const float*)d_in[8];   // entity_context_tab
  const float* rel_tab    = (const float*)d_in[9];
  const float* rel_gcn_w  = (const float*)d_in[10];
  const float* gate_e     = (const float*)d_in[11];
  const float* gate_r     = (const float*)d_in[12];
  const float* basis1     = (const float*)d_in[13];
  const float* att1       = (const float*)d_in[14];
  const float* root1      = (const float*)d_in[15];
  const float* bias1      = (const float*)d_in[16];
  const float* basis2     = (const float*)d_in[17];
  const float* att2       = (const float*)d_in[18];
  const float* root2      = (const float*)d_in[19];
  const float* bias2      = (const float*)d_in[20];
  float* out = (float*)d_out;

  // Workspace layout (bytes)
  char* ws = (char*)d_ws;
  const size_t WPACK_ELEMS = (size_t)KKT * KNCT * 32 * 16;            // 81920
  __bf16* WP1 = (__bf16*)(ws);                                        //    163,840 B
  __bf16* WP2 = (__bf16*)(ws + 163840);                               //    163,840 B
  __bf16* X0b = (__bf16*)(ws + 327680);                               // 12,800,000 B
  __bf16* X2b = (__bf16*)(ws + 327680 + 12800000ull);                 // 12,800,000 B
  float*  Y   = (float*)(ws + 327680 + 25600000ull);                  // 128,000,000 B
  float*  agg = (float*)(ws + 327680 + 153600000ull);                 //  25,600,000 B
  float*  cnt = (float*)(ws + 327680 + 179200000ull);                 //     200,000 B
  float*  X3  = (float*)(ws + 327680 + 179400000ull);                 //  25,600,000 B
  float*  T1  = (float*)(ws + 327680 + 205000000ull);                 //     102,400 B
  float*  RC  = (float*)(ws + 327680 + 205102400ull);                 //     102,400 B

  const int B = 256;
  const int gemmBlocks = (KMT * KNTG * 32) / B;   // 3125, exact -> EXEC all-ones
  const int packBlocks = (int)((WPACK_ELEMS + B - 1) / B);
  const int aggN       = KN_ENT * KDIM + KN_ENT;  // agg + cnt contiguous
  const int zeroBlocks = (aggN + B - 1) / B;
  const int edgeBlocks = (KNE * 32) / B;
  const int nodBlocks  = (KN_ENT * KDIM) / B;
  const int relBlocks  = (KNREL * KDIM + B - 1) / B;
  const int outBlocks  = (KNS * KDIM) / B;

  // Pack weights, convert layer-1 input to bf16 fragment feed
  dkge_pack_w<<<packBlocks, B, 0, stream>>>(basis1, root1, WP1);
  dkge_pack_w<<<packBlocks, B, 0, stream>>>(basis2, root2, WP2);
  dkge_cvt_bf16<<<nodBlocks, B, 0, stream>>>(ent_ctx0, X0b, KN_ENT * KDIM);

  // ---- Layer 1 ----
  dkge_zero<<<zeroBlocks, B, 0, stream>>>(agg, aggN);
  dkge_gemm_wmma<<<gemmBlocks, B, 0, stream>>>(X0b, WP1, Y);
  dkge_edge_scatter<<<edgeBlocks, B, 0, stream>>>(edge_index, edge_type, edge_norm,
                                                  att1, Y, agg, cnt);
  dkge_finalize_bf16<<<nodBlocks, B, 0, stream>>>(agg, cnt, Y, bias1, X2b);

  // ---- Layer 2 ----
  dkge_zero<<<zeroBlocks, B, 0, stream>>>(agg, aggN);
  dkge_gemm_wmma<<<gemmBlocks, B, 0, stream>>>(X2b, WP2, Y);
  dkge_edge_scatter<<<edgeBlocks, B, 0, stream>>>(edge_index, edge_type, edge_norm,
                                                  att2, Y, agg, cnt);
  dkge_finalize_f32<<<nodBlocks, B, 0, stream>>>(agg, cnt, Y, bias2, X3);

  // ---- Relation GCN ----
  dkge_rel_stage1<<<relBlocks, B, 0, stream>>>(DAD_rel, rel_tab, T1);
  dkge_rel_stage2<<<relBlocks, B, 0, stream>>>(T1, rel_gcn_w, RC);

  // ---- Gated gather output ----
  dkge_output<<<outBlocks, B, 0, stream>>>(samples, ent_emb, rel_emb, X3, RC,
                                           gate_e, gate_r, out);
}
